// InstanceConsistencyLoss_44710609551553
// MI455X (gfx1250) — matrix operand order repaired and checked by reading, not compiled, not run on hardware
//
#include <hip/hip_runtime.h>
#include <hip/hip_bf16.h>

typedef _Float16 v8h  __attribute__((ext_vector_type(8)));
typedef _Float16 v16h __attribute__((ext_vector_type(16)));
typedef float    v8f  __attribute__((ext_vector_type(8)));

#define NPIX 65536   // 256*256
#define NCH  128
#define NSEG 256
#define NB   8

// ---------------------------------------------------------------- zero
__global__ __launch_bounds__(256) void icl_zero(float* __restrict__ p, int n) {
    int i = blockIdx.x * 256 + threadIdx.x;
    if (i < n) p[i] = 0.0f;
}

// ---------------------------------------------------------------- pass 1: per-pixel norm + counts/sumsq
// grid: B * (NPIX/1024) = 512 blocks, 256 threads, 4 pixels per thread
__global__ __launch_bounds__(256) void icl_norm(const float* __restrict__ feat,
                                                const int*   __restrict__ gt,
                                                float* __restrict__ rinv_g,
                                                float* __restrict__ counts,
                                                float* __restrict__ sumsq) {
    __shared__ float scnt[NSEG];
    __shared__ float ssum[NSEG];
    const int tid = threadIdx.x;
    scnt[tid] = 0.0f; ssum[tid] = 0.0f;
    __syncthreads();

    const int b  = blockIdx.x >> 6;                 // 64 blocks per image
    const int p0 = ((blockIdx.x & 63) << 10) + (tid << 2);
    const size_t fbase = ((size_t)b << 23) + (size_t)p0;   // b*128*65536 + p0

    float4 acc = make_float4(0.f, 0.f, 0.f, 0.f);
    for (int c = 0; c < NCH; ++c) {
        float4 x = *(const float4*)(feat + fbase + ((size_t)c << 16));
        acc.x += x.x * x.x; acc.y += x.y * x.y;
        acc.z += x.z * x.z; acc.w += x.w * x.w;
    }
    float4 r;
    r.x = 1.0f / fmaxf(sqrtf(acc.x), 1e-12f);
    r.y = 1.0f / fmaxf(sqrtf(acc.y), 1e-12f);
    r.z = 1.0f / fmaxf(sqrtf(acc.z), 1e-12f);
    r.w = 1.0f / fmaxf(sqrtf(acc.w), 1e-12f);
    *(float4*)(rinv_g + ((size_t)b << 16) + p0) = r;

    // per-pixel sum_c f^2 = norm2 * rinv^2  (== 1 unless norm < eps)
    float qx = acc.x * r.x * r.x, qy = acc.y * r.y * r.y;
    float qz = acc.z * r.z * r.z, qw = acc.w * r.w * r.w;

    int4 id = *(const int4*)(gt + ((size_t)b << 16) + p0);
    atomicAdd(&scnt[id.x], 1.0f); atomicAdd(&ssum[id.x], qx);
    atomicAdd(&scnt[id.y], 1.0f); atomicAdd(&ssum[id.y], qy);
    atomicAdd(&scnt[id.z], 1.0f); atomicAdd(&ssum[id.z], qz);
    atomicAdd(&scnt[id.w], 1.0f); atomicAdd(&ssum[id.w], qw);
    __syncthreads();

    unsafeAtomicAdd(&counts[b * NSEG + tid], scnt[tid]);
    unsafeAtomicAdd(&sumsq [b * NSEG + tid], ssum[tid]);
}

// ---------------------------------------------------------------- pass 2: segment sums via one-hot WMMA GEMM
// grid: B*32 = 256 blocks (2048-pixel slab each, 64 chunks), 256 threads (8 waves)
// wave w owns segments [32w, 32w+32): 2 A-tiles x 8 channel B-tiles of f32 accumulators.
__global__ __launch_bounds__(256) void icl_segsum(const float* __restrict__ feat,
                                                  const int*   __restrict__ gt,
                                                  const float* __restrict__ rinv_g,
                                                  float* __restrict__ sums) {
    __shared__ _Float16 fhs[NCH * 40];   // [c][k] chunk, row stride 40 halves (80B) -> conflict-free b128
    __shared__ int ishare[32];

    const int tid  = threadIdx.x;
    const int lane = tid & 31;
    const int w    = tid >> 5;
    const int lh   = lane & 15;
    const int kq   = lane >> 4;          // 0: K-low half, 1: K-high half
    const int kb   = kq * 8;             // A-layout K base per lane group
    const int b    = blockIdx.x >> 5;
    const int P0   = (blockIdx.x & 31) * 2048;
    const int segbase = w * 32;

    const int cc = tid >> 1;             // channel handled in cooperative load
    const int k0 = (tid & 1) << 4;       // 0 or 16
    const size_t fbase = (((size_t)b * NCH + cc) << 16);

    v8f acc[2][8];
    const v8f vz = {};
#pragma unroll
    for (int st = 0; st < 2; ++st)
#pragma unroll
        for (int ct = 0; ct < 8; ++ct) acc[st][ct] = vz;
    bool touched0 = false, touched1 = false;

    for (int chunk = 0; chunk < 64; ++chunk) {
        const int p0 = P0 + chunk * 32;
        __syncthreads();   // protect LDS from prior readers
        if (tid < 32) ishare[tid] = gt[((size_t)b << 16) + p0 + tid];
        {
            const float4* fs = (const float4*)(feat   + fbase + p0 + k0);
            const float4* rs = (const float4*)(rinv_g + ((size_t)b << 16) + p0 + k0);
            float4 x0 = fs[0], x1 = fs[1], x2 = fs[2], x3 = fs[3];
            float4 r0 = rs[0], r1 = rs[1], r2 = rs[2], r3 = rs[3];
            v8h h0, h1;
            h0[0] = (_Float16)(x0.x * r0.x); h0[1] = (_Float16)(x0.y * r0.y);
            h0[2] = (_Float16)(x0.z * r0.z); h0[3] = (_Float16)(x0.w * r0.w);
            h0[4] = (_Float16)(x1.x * r1.x); h0[5] = (_Float16)(x1.y * r1.y);
            h0[6] = (_Float16)(x1.z * r1.z); h0[7] = (_Float16)(x1.w * r1.w);
            h1[0] = (_Float16)(x2.x * r2.x); h1[1] = (_Float16)(x2.y * r2.y);
            h1[2] = (_Float16)(x2.z * r2.z); h1[3] = (_Float16)(x2.w * r2.w);
            h1[4] = (_Float16)(x3.x * r3.x); h1[5] = (_Float16)(x3.y * r3.y);
            h1[6] = (_Float16)(x3.z * r3.z); h1[7] = (_Float16)(x3.w * r3.w);
            *(v8h*)(&fhs[cc * 40 + k0])     = h0;
            *(v8h*)(&fhs[cc * 40 + k0 + 8]) = h1;
        }
        __syncthreads();

        // per-lane ids following the f16 A 16x32 K layout
        const int4* i4 = (const int4*)ishare;
        const int kb4 = kq * 2;
        int4 ia = i4[kb4], ib2 = i4[kb4 + 1], ic = i4[kb4 + 4], id2 = i4[kb4 + 5];
        int idv[16] = { ia.x, ia.y, ia.z, ia.w,  ib2.x, ib2.y, ib2.z, ib2.w,
                        ic.x, ic.y, ic.z, ic.w,  id2.x, id2.y, id2.z, id2.w };
        bool m0 = false, m1 = false;
#pragma unroll
        for (int e = 0; e < 16; ++e) {
            unsigned d = (unsigned)(idv[e] - segbase);
            m0 |= (d < 16u);
            m1 |= (d >= 16u) & (d < 32u);
        }
        const bool a0 = __any((int)m0) != 0;   // wave-uniform
        const bool a1 = __any((int)m1) != 0;
        if (!(a0 | a1)) continue;              // all barriers are outside this region
        touched0 |= a0; touched1 |= a1;

        v16h A0, A1;
        const int seg0 = segbase + lh, seg1 = seg0 + 16;
#pragma unroll
        for (int e = 0; e < 16; ++e) {
            A0[e] = (idv[e] == seg0) ? (_Float16)1.0f : (_Float16)0.0f;
            A1[e] = (idv[e] == seg1) ? (_Float16)1.0f : (_Float16)0.0f;
        }
#pragma unroll
        for (int ct = 0; ct < 8; ++ct) {
            const _Float16* bp = &fhs[(ct * 16 + lh) * 40 + kb];
            v8h blo = *(const v8h*)bp;
            v8h bhi = *(const v8h*)(bp + 16);
            v16h Bv = __builtin_shufflevector(blo, bhi, 0,1,2,3,4,5,6,7,8,9,10,11,12,13,14,15);
            if (a0) acc[0][ct] = __builtin_amdgcn_wmma_f32_16x16x32_f16(
                                     false, A0, false, Bv, (short)0, acc[0][ct], false, false);
            if (a1) acc[1][ct] = __builtin_amdgcn_wmma_f32_16x16x32_f16(
                                     false, A1, false, Bv, (short)0, acc[1][ct], false, false);
        }
    }

    // write back partial sums (C/D layout: lanes 0-15 -> M=r, lanes 16-31 -> M=r+8; N=lane&15)
#pragma unroll
    for (int st = 0; st < 2; ++st) {
        if (st == 0 ? !touched0 : !touched1) continue;
#pragma unroll
        for (int ct = 0; ct < 8; ++ct) {
#pragma unroll
            for (int r = 0; r < 8; ++r) {
                int seg = segbase + st * 16 + r + kq * 8;
                int chn = ct * 16 + lh;
                unsafeAtomicAdd(&sums[(((size_t)b * NSEG + seg) << 7) + chn], acc[st][ct][r]);
            }
        }
    }
}

// ---------------------------------------------------------------- pass 3: per-image loss (centroids + WMMA Gram)
// grid: B blocks, 256 threads (8 waves)
__global__ __launch_bounds__(256) void icl_finalize(const float* __restrict__ sums,
                                                    const float* __restrict__ counts,
                                                    const float* __restrict__ sumsq,
                                                    _Float16* __restrict__ centh,
                                                    float* __restrict__ losses) {
    __shared__ float cn2s[NSEG];
    __shared__ float vfs[NSEG];
    __shared__ float red[3];   // 0: compact sum, 1: nv, 2: sep sum
    const int tid = threadIdx.x;
    const int b = blockIdx.x;
    if (tid < 3) red[tid] = 0.0f;
    __syncthreads();

    {   // phase 1: centroids, cn2, validity, compactness
        const int s = tid;
        float cnt  = counts[b * NSEG + s];
        float ssq  = sumsq [b * NSEG + s];
        float safe = fmaxf(cnt, 1.0f);
        float inv  = 1.0f / safe;
        const float* srow = sums  + (((size_t)b * NSEG + s) << 7);
        _Float16*    crow = centh + (((size_t)b * NSEG + s) << 7);
        float c2 = 0.0f;
        for (int ch = 0; ch < NCH; ++ch) {
            float v = srow[ch] * inv;
            c2 += v * v;
            crow[ch] = (_Float16)v;
        }
        cn2s[s] = c2;
        bool valid = (cnt >= 4.0f) && (s != 0) && (s != 255);
        vfs[s] = valid ? 1.0f : 0.0f;
        if (valid) {
            float compact = (ssq - c2 * cnt) / (128.0f * safe);
            atomicAdd(&red[0], compact);
            atomicAdd(&red[1], 1.0f);
        }
    }
    __threadfence();
    __syncthreads();

    // phase 2: Gram = cent @ cent^T via WMMA; 256 16x16 tiles, 32 per wave
    const int lane = tid & 31, w = tid >> 5;
    const int lh = lane & 15, kq = lane >> 4, kb = kq * 8;
    float sep = 0.0f;
    for (int t = 0; t < 32; ++t) {
        const int tile = w * 32 + t;
        const int ti = tile >> 4, tj = tile & 15;
        const _Float16* arow = centh + (((size_t)b * NSEG + ti * 16 + lh) << 7);
        const _Float16* brow = centh + (((size_t)b * NSEG + tj * 16 + lh) << 7);
        v8f acc = {};
#pragma unroll
        for (int kc = 0; kc < 4; ++kc) {
            v8h alo = *(const v8h*)(arow + kc * 32 + kb);
            v8h ahi = *(const v8h*)(arow + kc * 32 + 16 + kb);
            v16h A = __builtin_shufflevector(alo, ahi, 0,1,2,3,4,5,6,7,8,9,10,11,12,13,14,15);
            v8h blo = *(const v8h*)(brow + kc * 32 + kb);
            v8h bhi = *(const v8h*)(brow + kc * 32 + 16 + kb);
            v16h B = __builtin_shufflevector(blo, bhi, 0,1,2,3,4,5,6,7,8,9,10,11,12,13,14,15);
            acc = __builtin_amdgcn_wmma_f32_16x16x32_f16(false, A, false, B, (short)0, acc, false, false);
        }
#pragma unroll
        for (int r = 0; r < 8; ++r) {
            int i = ti * 16 + r + kq * 8;
            int j = tj * 16 + lh;
            float g  = acc[r];
            float d2 = fmaxf(cn2s[i] + cn2s[j] - 2.0f * g, 0.0f);
            float pr = vfs[i] * vfs[j] * ((i == j) ? 0.0f : 1.0f);
            sep += expf(-d2) * pr;
        }
    }
    atomicAdd(&red[2], sep);
    __syncthreads();

    if (tid == 0) {
        float nv = red[1];
        float Lc = (nv > 0.0f) ? red[0] / fmaxf(nv, 1.0f) : 0.0f;
        float Ls = (nv > 1.0f) ? red[2] / fmaxf(nv * nv, 1.0f) : 0.0f;
        losses[b] = (nv > 0.0f) ? (Lc + 2.0f * Ls) : 0.0f;
    }
}

// ---------------------------------------------------------------- final deterministic reduce
__global__ void icl_reduce(const float* __restrict__ losses, float* __restrict__ out) {
    if (threadIdx.x == 0 && blockIdx.x == 0) {
        float s = 0.0f;
        for (int i = 0; i < NB; ++i) s += losses[i];
        out[0] = s * 0.125f;
    }
}

// ---------------------------------------------------------------- launch
extern "C" void kernel_launch(void* const* d_in, const int* in_sizes, int n_in,
                              void* d_out, int out_size, void* d_ws, size_t ws_size,
                              hipStream_t stream) {
    const float* feat = (const float*)d_in[0];     // (8,128,256,256) f32
    const int*   gt   = (const int*)d_in[1];       // (8,256,256) i32
    float* out = (float*)d_out;

    float* ws     = (float*)d_ws;
    float* rinv   = ws;                            // 524288 f32
    float* sums   = ws + 524288;                   // 262144 f32
    float* counts = ws + 786432;                   // 2048 f32
    float* sumsq  = ws + 788480;                   // 2048 f32
    float* losses = ws + 790528;                   // 8 f32
    _Float16* centh = (_Float16*)(ws + 790536);    // 262144 f16 (16B-aligned offset)

    // zero sums+counts+sumsq (contiguous 266240 f32)
    icl_zero<<<(266240 + 255) / 256, 256, 0, stream>>>(sums, 266240);
    icl_norm<<<NB * (NPIX / 1024), 256, 0, stream>>>(feat, gt, rinv, counts, sumsq);
    icl_segsum<<<NB * 32, 256, 0, stream>>>(feat, gt, rinv, sums);
    icl_finalize<<<NB, 256, 0, stream>>>(sums, counts, sumsq, centh, losses);
    icl_reduce<<<1, 32, 0, stream>>>(losses, out);
}
